// MoE_33423435498014
// MI455X (gfx1250) — matrix operand order repaired and checked by reading, not compiled, not run on hardware
//
#include <hip/hip_runtime.h>
#include <hip/hip_bf16.h>
#include <math.h>

// Problem constants (match reference)
#define T_TOK 2048
#define D_IN  2048
#define H_IM  1024     // per-expert intermediate; shared SH == H_IM
#define E_EXP 16
#define K_TOP 4

// Tiling
#define BM 64          // tokens per workgroup tile
#define BK 32          // WMMA K step (bf16 16x16x32)
#define BN 64          // N tile (stage1: of H, stage2: of D)

typedef __attribute__((ext_vector_type(16))) __bf16          v16bf;
typedef __attribute__((ext_vector_type(16))) unsigned short  us16;
typedef __attribute__((ext_vector_type(8)))  unsigned short  us8;
typedef __attribute__((ext_vector_type(8)))  float           v8f;
typedef __attribute__((ext_vector_type(4)))  float           f4;

__device__ __forceinline__ unsigned short f2bf(float f) {
  // native f32->bf16 (RNE); lowers to v_cvt_pk_bf16_f32 on gfx1250
  __bf16 b = (__bf16)f;
  return __builtin_bit_cast(unsigned short, b);
}

__device__ __forceinline__ us8 cvt8(f4 a, f4 b) {
  us8 o;
  o[0] = f2bf(a[0]); o[1] = f2bf(a[1]); o[2] = f2bf(a[2]); o[3] = f2bf(a[3]);
  o[4] = f2bf(b[0]); o[5] = f2bf(b[1]); o[6] = f2bf(b[2]); o[7] = f2bf(b[3]);
  return o;
}

// ---------------------------------------------------------------------------
// Kernel 1: gate scores -> softmax -> top-4 -> dense per-token weights [T,E]
// ---------------------------------------------------------------------------
__global__ __launch_bounds__(256) void gate_topk_kernel(
    const float* __restrict__ x, const float* __restrict__ gw,
    float* __restrict__ tok_w) {
  __shared__ float red[E_EXP][256];
  const int t = blockIdx.x;
  const int tid = threadIdx.x;
  float acc[E_EXP];
#pragma unroll
  for (int e = 0; e < E_EXP; e++) acc[e] = 0.f;
  for (int d = tid; d < D_IN; d += 256) {
    float xv = x[(size_t)t * D_IN + d];
#pragma unroll
    for (int e = 0; e < E_EXP; e++) acc[e] += xv * gw[(size_t)e * D_IN + d];
  }
#pragma unroll
  for (int e = 0; e < E_EXP; e++) red[e][tid] = acc[e];
  __syncthreads();
  for (int s = 128; s > 0; s >>= 1) {
    if (tid < s) {
#pragma unroll
      for (int e = 0; e < E_EXP; e++) red[e][tid] += red[e][tid + s];
    }
    __syncthreads();
  }
  if (tid == 0) {
    float sc[E_EXP];
    float mx = -1e30f;
    for (int e = 0; e < E_EXP; e++) { sc[e] = red[e][0]; mx = fmaxf(mx, sc[e]); }
    float sum = 0.f;
    for (int e = 0; e < E_EXP; e++) { sc[e] = __expf(sc[e] - mx); sum += sc[e]; }
    float inv = 1.f / sum;
    for (int e = 0; e < E_EXP; e++) sc[e] *= inv;
    float outw[E_EXP];
    bool used[E_EXP];
    for (int e = 0; e < E_EXP; e++) { outw[e] = 0.f; used[e] = false; }
    for (int k = 0; k < K_TOP; k++) {           // route_scale = 1.0, no renorm
      int bi = 0; float bv = -1.f;
      for (int e = 0; e < E_EXP; e++)
        if (!used[e] && sc[e] > bv) { bv = sc[e]; bi = e; }
      used[bi] = true; outw[bi] = bv;
    }
    for (int e = 0; e < E_EXP; e++) tok_w[(size_t)t * E_EXP + e] = outw[e];
  }
}

// ---------------------------------------------------------------------------
// Kernel 2: deterministic per-expert token gather lists (token order preserved)
// ---------------------------------------------------------------------------
__global__ void build_lists_kernel(const float* __restrict__ tok_w,
                                   int* __restrict__ counts,
                                   int* __restrict__ tok_list,
                                   float* __restrict__ wgt_list) {
  int e = threadIdx.x;
  if (e >= E_EXP) return;
  int cnt = 0;
  for (int t = 0; t < T_TOK; t++) {
    float w = tok_w[(size_t)t * E_EXP + e];
    if (w > 0.f) {
      tok_list[e * T_TOK + cnt] = t;
      wgt_list[e * T_TOK + cnt] = w;
      cnt++;
    }
  }
  counts[e] = cnt;
}

// ---------------------------------------------------------------------------
// Kernel 3: fused gated-MLP expert. SHARED=true: identity tokens, weight 1,
// plain stores (initializes out). SHARED=false: gathered tokens, atomic adds.
// bf16 WMMA, f32 accumulate, double-buffered LDS staging (1 barrier / chunk).
// ---------------------------------------------------------------------------
template <bool SHARED>
__global__ __launch_bounds__(256) void moe_expert_kernel(
    const float* __restrict__ x,
    const float* __restrict__ w1,   // routed: [E][2H][D]; shared: [2H][D]
    const float* __restrict__ w2,   // routed: [E][D][H];  shared: [D][H]
    const int* __restrict__ counts,
    const int* __restrict__ tok_list,
    const float* __restrict__ wgt_list,
    float* __restrict__ out) {
  __shared__ unsigned short hbuf[BM * H_IM];     // 128 KB: full h tile in LDS
  __shared__ unsigned short xs[2][BM * BK];      // 2x4 KB double-buffered x
  __shared__ unsigned short wyb[2][BN * BK];     // 2x4 KB w1(y) / w2
  __shared__ unsigned short wgb[2][BN * BK];     // 2x4 KB w1(gate)
  __shared__ float twgt[BM];
  __shared__ int   tids[BM];

  const int e     = SHARED ? 0 : blockIdx.y;
  const int tile  = blockIdx.x;
  const int count = SHARED ? T_TOK : counts[e];
  if (tile * BM >= count) return;                // block-uniform exit

  const int tid   = threadIdx.x;
  const int wave  = tid >> 5;                    // 8 waves of 32 (wave32)
  const int lane  = tid & 31;
  const int strip = wave & 3;                    // 16-row strip 0..3
  const int nhalf = wave >> 2;                   // columns 0..31 / 32..63
  const int lrow  = lane & 15;                   // A: row-in-strip; B/C: column
  const int hiK   = lane >> 4;                   // half-wave K split
  const int srow  = tid >> 2;                    // staging: row this thread fills
  const int scol  = (tid & 3) * 8;               // staging: first of 8 columns

  const float* w1e = SHARED ? w1 : (w1 + (size_t)e * (2 * H_IM) * D_IN);
  const float* w2e = SHARED ? w2 : (w2 + (size_t)e * D_IN * H_IM);

  if (tid < BM) {
    int slot = tile * BM + tid;
    if (SHARED)            { tids[tid] = slot;                        twgt[tid] = 1.f; }
    else if (slot < count) { tids[tid] = tok_list[e * T_TOK + slot];  twgt[tid] = wgt_list[e * T_TOK + slot]; }
    else                   { tids[tid] = 0;                           twgt[tid] = 0.f; }  // pad rows
  }
  __syncthreads();
  const size_t xrow = (size_t)tids[srow] * D_IN + scol;  // gathered token row

  // ---------------- Stage 1: h = y * silu(gate), kept in LDS as bf16 --------
#pragma unroll 1
  for (int n1 = 0; n1 < H_IM; n1 += BN) {
    v8f accY0 = {}, accY1 = {}, accG0 = {}, accG1 = {};
    const float* pyrow = &w1e[(size_t)(n1 + srow) * D_IN + scol];
    const float* pgrow = &w1e[(size_t)(H_IM + n1 + srow) * D_IN + scol];
    // prologue: fetch chunk 0 into registers
    f4 lx0 = *(const f4*)&x[xrow], lx1 = *(const f4*)&x[xrow + 4];
    f4 ly0 = *(const f4*)pyrow,    ly1 = *(const f4*)(pyrow + 4);
    f4 lg0 = *(const f4*)pgrow,    lg1 = *(const f4*)(pgrow + 4);
    int p = 0;
#pragma unroll 1
    for (int kd = 0; kd < D_IN; kd += BK) {
      // commit in-flight chunk to LDS buffer p (one ds_store_b128 each)
      *(us8*)&xs[p][srow * BK + scol]  = cvt8(lx0, lx1);
      *(us8*)&wyb[p][srow * BK + scol] = cvt8(ly0, ly1);
      *(us8*)&wgb[p][srow * BK + scol] = cvt8(lg0, lg1);
      __syncthreads();
      if (kd + BK < D_IN) {                      // issue next chunk's loads now
        int kn = kd + BK;
        lx0 = *(const f4*)&x[xrow + kn];   lx1 = *(const f4*)&x[xrow + kn + 4];
        ly0 = *(const f4*)(pyrow + kn);    ly1 = *(const f4*)(pyrow + kn + 4);
        lg0 = *(const f4*)(pgrow + kn);    lg1 = *(const f4*)(pgrow + kn + 4);
      }
      // A fragment: 16x32 bf16 in contiguous K order (K = hiK*16 + j) (row=strip*16+lrow, K split)
      us16 au;
      {
        const unsigned short* xr = &xs[p][(strip * 16 + lrow) * BK];
        int kb = hiK * 16;
#pragma unroll
        for (int j = 0; j < 16; j++) au[j] = xr[kb + j];
      }
      v16bf a = __builtin_bit_cast(v16bf, au);
      // B fragments: lane col = base+lrow, K = hiK*16 + j
      us16 b0, b1, b2, b3;
      {
        int kb = hiK * 16;
        const unsigned short* py0 = &wyb[p][(nhalf * 32 + 0  + lrow) * BK + kb];
        const unsigned short* py1 = &wyb[p][(nhalf * 32 + 16 + lrow) * BK + kb];
        const unsigned short* pg0 = &wgb[p][(nhalf * 32 + 0  + lrow) * BK + kb];
        const unsigned short* pg1 = &wgb[p][(nhalf * 32 + 16 + lrow) * BK + kb];
#pragma unroll
        for (int j = 0; j < 16; j++) { b0[j]=py0[j]; b1[j]=py1[j]; b2[j]=pg0[j]; b3[j]=pg1[j]; }
      }
      accY0 = __builtin_amdgcn_wmma_f32_16x16x32_bf16(false, a, false, __builtin_bit_cast(v16bf, b0), (short)0, accY0, false, false);
      accY1 = __builtin_amdgcn_wmma_f32_16x16x32_bf16(false, a, false, __builtin_bit_cast(v16bf, b1), (short)0, accY1, false, false);
      accG0 = __builtin_amdgcn_wmma_f32_16x16x32_bf16(false, a, false, __builtin_bit_cast(v16bf, b2), (short)0, accG0, false, false);
      accG1 = __builtin_amdgcn_wmma_f32_16x16x32_bf16(false, a, false, __builtin_bit_cast(v16bf, b3), (short)0, accG1, false, false);
      p ^= 1;
    }
    // SiLU gating: y * g * rcp(1+exp(-g)) -- v_exp_f32 + v_rcp_f32, no IEEE div
#pragma unroll
    for (int r = 0; r < 8; r++) {
      int m    = strip * 16 + (hiK ? 8 : 0) + r;   // C/D layout: VGPR r -> M
      int ncol = n1 + nhalf * 32 + lrow;
      float g0 = accG0[r], g1 = accG1[r];
      float h0 = accY0[r] * g0 * __builtin_amdgcn_rcpf(1.f + __expf(-g0));
      float h1 = accY1[r] * g1 * __builtin_amdgcn_rcpf(1.f + __expf(-g1));
      hbuf[m * H_IM + ncol]      = f2bf(h0);
      hbuf[m * H_IM + ncol + 16] = f2bf(h1);
    }
    __syncthreads();                               // hbuf/x-stage quiesce
  }

  // ---------------- Stage 2: out (+)= (h @ w2^T) * token_weight -------------
#pragma unroll 1
  for (int n2 = 0; n2 < D_IN; n2 += BN) {
    v8f acc0 = {}, acc1 = {};
    const float* pwrow = &w2e[(size_t)(n2 + srow) * H_IM + scol];
    f4 lw0 = *(const f4*)pwrow, lw1 = *(const f4*)(pwrow + 4);
    int p = 0;
#pragma unroll 1
    for (int kh = 0; kh < H_IM; kh += BK) {
      *(us8*)&wyb[p][srow * BK + scol] = cvt8(lw0, lw1);
      __syncthreads();
      if (kh + BK < H_IM) {
        int kn = kh + BK;
        lw0 = *(const f4*)(pwrow + kn); lw1 = *(const f4*)(pwrow + kn + 4);
      }
      us16 au;
      {
        const unsigned short* hr = &hbuf[(strip * 16 + lrow) * H_IM + kh];
        int kb = hiK * 16;
#pragma unroll
        for (int j = 0; j < 16; j++) au[j] = hr[kb + j];
      }
      v16bf a = __builtin_bit_cast(v16bf, au);
      us16 b0, b1;
      {
        int kb = hiK * 16;
        const unsigned short* p0 = &wyb[p][(nhalf * 32 + 0  + lrow) * BK + kb];
        const unsigned short* p1 = &wyb[p][(nhalf * 32 + 16 + lrow) * BK + kb];
#pragma unroll
        for (int j = 0; j < 16; j++) { b0[j] = p0[j]; b1[j] = p1[j]; }
      }
      acc0 = __builtin_amdgcn_wmma_f32_16x16x32_bf16(false, a, false, __builtin_bit_cast(v16bf, b0), (short)0, acc0, false, false);
      acc1 = __builtin_amdgcn_wmma_f32_16x16x32_bf16(false, a, false, __builtin_bit_cast(v16bf, b1), (short)0, acc1, false, false);
      p ^= 1;
    }
#pragma unroll
    for (int r = 0; r < 8; r++) {
      int m     = strip * 16 + (hiK ? 8 : 0) + r;
      int tokid = tids[m];
      float wt  = twgt[m];
      int c0    = n2 + nhalf * 32 + lrow;
      float v0 = acc0[r] * wt, v1 = acc1[r] * wt;
      float* o = out + (size_t)tokid * D_IN + c0;
      if (SHARED) { o[0] = v0; o[16] = v1; }                    // init out
      else        { atomicAdd(o, v0); atomicAdd(o + 16, v1); }  // f32 atomic add
    }
    __syncthreads();                               // wyb quiesce before reuse
  }
}

// ---------------------------------------------------------------------------
extern "C" void kernel_launch(void* const* d_in, const int* in_sizes, int n_in,
                              void* d_out, int out_size, void* d_ws, size_t ws_size,
                              hipStream_t stream) {
  const float* x   = (const float*)d_in[0];
  const float* gw  = (const float*)d_in[1];
  const float* w1  = (const float*)d_in[2];
  const float* w2  = (const float*)d_in[3];
  const float* sw1 = (const float*)d_in[4];
  const float* sw2 = (const float*)d_in[5];
  float* out = (float*)d_out;

  char* ws = (char*)d_ws;
  float* tok_w    = (float*)ws;  ws += (size_t)T_TOK * E_EXP * sizeof(float);
  int*   counts   = (int*)ws;    ws += 256;  // 64B used, keep alignment
  int*   tok_list = (int*)ws;    ws += (size_t)E_EXP * T_TOK * sizeof(int);
  float* wgt_list = (float*)ws;  ws += (size_t)E_EXP * T_TOK * sizeof(float);

  gate_topk_kernel<<<T_TOK, 256, 0, stream>>>(x, gw, tok_w);
  build_lists_kernel<<<1, 32, 0, stream>>>(tok_w, counts, tok_list, wgt_list);
  // Shared expert first: plain stores initialize every out element.
  moe_expert_kernel<true><<<dim3(T_TOK / BM, 1), 256, 0, stream>>>(
      x, sw1, sw2, nullptr, nullptr, nullptr, out);
  // Routed experts: atomic accumulation on top. Stream order serializes.
  moe_expert_kernel<false><<<dim3(T_TOK / BM, E_EXP), 256, 0, stream>>>(
      x, w1, w2, counts, tok_list, wgt_list, out);
}
